// HandwritingSynthesis_47751446397533
// MI455X (gfx1250) — matrix operand order repaired and checked by reading, not compiled, not run on hardware
//
#include <hip/hip_runtime.h>
#include <hip/hip_bf16.h>
#include <math.h>

// ---------------- problem constants ----------------
constexpr int cB = 64, cT = 600, cS = 64, cH = 400, cK = 10, cMIX = 20;
constexpr int cSD = 60;                 // SENT_DIM
constexpr int cG  = 4 * cH;             // 1600 gate width
constexpr int cIN1P = 64;               // IN1=63 padded
constexpr int cHP   = 416;              // H=400 padded to 32-mult
constexpr int cIN2P = 480;              // IN2=463 padded
constexpr int cCATP = 1216;             // 3H=1200 padded
constexpr int cNT   = 100;              // 1600/16 gate n-tiles
constexpr int cKT1  = 2;                // 64/32
constexpr int cKTH  = 13;               // 416/32
constexpr int cKT2  = 15;               // 480/32
constexpr int cKTC  = 38;               // 1216/32
constexpr int NBLK  = 13;               // persistent blocks (104 waves, 100 active)
constexpr int cE    = cB * cT;          // 38400

// ---------------- workspace layout (bytes, all 256-aligned) ----------------
constexpr size_t O_BAR   = 0;                         // 3 barrier counters
constexpr size_t O_SENT  = 256;                       // 64 f32
constexpr size_t O_KAPPA = 512;                       // 2 x [64][10] f32
constexpr size_t O_P     = O_KAPPA + 5120;            // (spare)
constexpr size_t O_PHI   = O_P     + 7680;            // [64][65] f32
constexpr size_t O_WBUF  = O_PHI   + 16640;           // [64][60] f32
constexpr size_t O_C1    = O_WBUF  + 15360;           // [64][400] f32
constexpr size_t O_C2    = O_C1    + 102400;
constexpr size_t O_C3    = O_C2    + 102400;
constexpr size_t O_H1S   = O_C3    + 102400;          // h fp32 state (also feeds attention)
constexpr size_t O_H2S   = O_H1S   + 102400;
constexpr size_t O_H3S   = O_H2S   + 102400;
constexpr size_t O_HB1   = O_H3S   + 102400;          // 2 x [64][416] bf16 (parity)
constexpr size_t O_HB2   = O_HB1   + 106496;
constexpr size_t O_HB3   = O_HB2   + 106496;
constexpr size_t O_INP1  = O_HB3   + 106496;          // [64][64] bf16
constexpr size_t O_WIH1P = O_INP1  + 8192;            // packed bf16 weights
constexpr size_t O_WHH1P = O_WIH1P + 204800;
constexpr size_t O_WIH2P = O_WHH1P + 1331200;
constexpr size_t O_WHH2P = O_WIH2P + 1536000;
constexpr size_t O_WIH3P = O_WHH2P + 1331200;
constexpr size_t O_WHH3P = O_WIH3P + 1536000;
constexpr size_t O_WMDNP = O_WHH3P + 1331200;
constexpr size_t O_IN2   = O_WMDNP + 311296;          // [T][64][480] bf16
constexpr size_t O_IN3   = O_IN2   + 36864000ull;     // [T][64][480] bf16
constexpr size_t O_CAT   = O_IN3   + 36864000ull;     // [T][64][1216] bf16
constexpr size_t O_PAR   = O_CAT   + 93388800ull;     // [T][64][128] f32 raw MDN params
// total ~195 MB

// ---------------- WMMA helpers ----------------
typedef __attribute__((ext_vector_type(16))) __bf16 bf16x16;
typedef __attribute__((ext_vector_type(8)))  float  f32x8;

struct __align__(16) U128 { unsigned x, y, z, w; };
union FragU { U128 q[2]; bf16x16 v; };

__device__ __forceinline__ unsigned short f2bf(float f) {
  unsigned u = __builtin_bit_cast(unsigned, f);
  unsigned r = (u + 0x7FFFu + ((u >> 16) & 1u)) >> 16;   // RNE
  return (unsigned short)r;
}
__device__ __forceinline__ float sigm(float x) { return 1.f / (1.f + expf(-x)); }

// A fragment (16x32 bf16, ISA 7.12.2 layout) from row-major [rows][stride] bf16,
// stride multiple of 32 -> two aligned 16B loads per lane.
__device__ __forceinline__ bf16x16 load_afrag(const unsigned short* base, int stride,
                                              int mbase, int kt, int lane) {
  const int row = mbase + (lane & 15);
  const unsigned short* p0 = base + (size_t)row * stride + kt * 32 + ((lane & 16) ? 8 : 0);
  FragU u;
  u.q[0] = ((const U128*)p0)[0];   // K = c .. c+7
  u.q[1] = ((const U128*)p0)[2];   // K = c+16 .. c+23
  return u.v;
}
// B fragment from pre-packed weights: [tile][lane][16] contiguous
__device__ __forceinline__ bf16x16 load_bfrag(const unsigned short* pw, int tile, int lane) {
  const unsigned short* p0 = pw + ((size_t)tile << 9) + (lane << 4);
  FragU u;
  u.q[0] = ((const U128*)p0)[0];
  u.q[1] = ((const U128*)p0)[1];
  return u.v;
}
__device__ __forceinline__ f32x8 wmma_bf16(bf16x16 a, bf16x16 b, f32x8 c) {
  return __builtin_amdgcn_wmma_f32_16x16x32_bf16(false, a, false, b, (short)0, c, false, false);
}

// ---------------- device-wide barrier (monotonic ticket) ----------------
__device__ __forceinline__ void grid_barrier(unsigned* bar, unsigned nblk) {
  __syncthreads();
  if (threadIdx.x == 0) {
    __threadfence();
    unsigned old = atomicAdd(bar, 1u);
    unsigned target = (old / nblk + 1u) * nblk;
    volatile unsigned* vb = (volatile unsigned*)bar;
    while (*vb < target) { __builtin_amdgcn_s_sleep(2); }
    __threadfence();
  }
  __syncthreads();
}

// ---------------- init: states, x-columns, pads, barrier counters ----------------
__global__ __launch_bounds__(256) void hs_init(
    const float* __restrict__ strks, const float* __restrict__ sents_m,
    const float* __restrict__ w_prev, const float* __restrict__ k_prev,
    const float* __restrict__ h1i, const float* __restrict__ c1i,
    const float* __restrict__ h2i, const float* __restrict__ c2i,
    const float* __restrict__ h3i, const float* __restrict__ c3i, char* ws) {
  const int gid = blockIdx.x * 256 + threadIdx.x;       // 0..38399
  unsigned short* in2 = (unsigned short*)(ws + O_IN2);
  unsigned short* in3 = (unsigned short*)(ws + O_IN3);
  unsigned short* cat = (unsigned short*)(ws + O_CAT);
  {
    const int R = gid, t = R >> 6, b = R & 63;
    const size_t r480 = (size_t)R * cIN2P, r1216 = (size_t)R * cCATP;
    for (int i = 0; i < 3; ++i) {
      unsigned short v = f2bf(strks[((size_t)b * cT + t) * 3 + i]);
      in2[r480 + i] = v; in3[r480 + i] = v;
    }
    for (int i = 463; i < 480; ++i) { in2[r480 + i] = 0; in3[r480 + i] = 0; }
    for (int i = 1200; i < 1216; ++i) cat[r1216 + i] = 0;
  }
  if (gid == 0) {
    unsigned* bar = (unsigned*)(ws + O_BAR);
    bar[0] = 0; bar[1] = 0; bar[2] = 0;
  }
  if (gid < 64) {
    const int b = gid;
    float s = 0.f;
    for (int i = 0; i < cS; ++i) s += sents_m[b * cS + i];
    ((float*)(ws + O_SENT))[b] = s;
    unsigned short* inp1 = (unsigned short*)(ws + O_INP1);
    for (int i = 0; i < 3; ++i) inp1[b * 64 + i] = f2bf(strks[(size_t)b * cT * 3 + i]);
    for (int d = 0; d < cSD; ++d) inp1[b * 64 + 3 + d] = f2bf(w_prev[b * cSD + d]);
    inp1[b * 64 + 63] = 0;
  }
  if (gid < 640) ((float*)(ws + O_KAPPA))[gid] = k_prev[gid];
  if (gid < 25600) {
    ((float*)(ws + O_C1))[gid] = c1i[gid];
    ((float*)(ws + O_C2))[gid] = c2i[gid];
    ((float*)(ws + O_C3))[gid] = c3i[gid];
  }
  if (gid < 26624) {                                    // 64*416 per parity
    const int b = gid / cHP, j = gid % cHP;
    unsigned short v1 = (j < cH) ? f2bf(h1i[b * cH + j]) : (unsigned short)0;
    unsigned short v2 = (j < cH) ? f2bf(h2i[b * cH + j]) : (unsigned short)0;
    unsigned short v3 = (j < cH) ? f2bf(h3i[b * cH + j]) : (unsigned short)0;
    unsigned short* hb1 = (unsigned short*)(ws + O_HB1);
    unsigned short* hb2 = (unsigned short*)(ws + O_HB2);
    unsigned short* hb3 = (unsigned short*)(ws + O_HB3);
    hb1[gid] = v1; hb1[26624 + gid] = 0;
    hb2[gid] = v2; hb2[26624 + gid] = 0;
    hb3[gid] = v3; hb3[26624 + gid] = 0;
  }
}

// ---------------- weight pack: fp32 [N][K] -> bf16 B-fragment tiles ----------------
__global__ __launch_bounds__(256) void hs_pack(
    const float* __restrict__ W, unsigned short* __restrict__ dst,
    int Ndim, int Kdim, int NT, int KT) {
  const int idx = blockIdx.x * 256 + threadIdx.x;
  const int total = NT * KT * 512;
  if (idx >= total) return;
  const int e = idx & 15, lane = (idx >> 4) & 31, tile = idx >> 9;
  const int kt = tile % KT, nt = tile / KT;
  const int N = nt * 16 + (lane & 15);
  const int Kk = kt * 32 + (e & 7) + ((lane >> 4) << 3) + ((e >> 3) << 4);      // B layout: N striped, K table order (8h+s, 16+8h+s-8)
  float v = (N < Ndim && Kk < Kdim) ? W[(size_t)N * Kdim + Kk] : 0.f;
  dst[idx] = f2bf(v);
}

// ---------------- phase 1: LSTM1 + attention window (persistent) ----------------
// 2 grid barriers per step: [gates GEMM + LSTM pointwise] -> barrier ->
// [per-batch-row window projection + attention + next-input build] -> barrier.
__global__ __launch_bounds__(256) void hs_phase1(
    const float* __restrict__ strks, const float* __restrict__ onehots,
    const float* __restrict__ bih1, const float* __restrict__ bhh1,
    const float* __restrict__ Wwin, const float* __restrict__ bwin, char* ws) {
  unsigned* bar = (unsigned*)(ws + O_BAR);
  float* sentlen = (float*)(ws + O_SENT);
  float* kappa   = (float*)(ws + O_KAPPA);
  float* phibuf  = (float*)(ws + O_PHI);
  float* wbuf    = (float*)(ws + O_WBUF);
  float* c1s     = (float*)(ws + O_C1);
  float* h1s     = (float*)(ws + O_H1S);
  unsigned short* hb0  = (unsigned short*)(ws + O_HB1);
  unsigned short* hb1p = hb0 + 26624;
  unsigned short* inp1 = (unsigned short*)(ws + O_INP1);
  unsigned short* in2  = (unsigned short*)(ws + O_IN2);
  unsigned short* in3  = (unsigned short*)(ws + O_IN3);
  unsigned short* cat  = (unsigned short*)(ws + O_CAT);
  const unsigned short* Wih1p = (const unsigned short*)(ws + O_WIH1P);
  const unsigned short* Whh1p = (const unsigned short*)(ws + O_WHH1P);

  const int lane  = threadIdx.x & 31;
  const int gwave = blockIdx.x * 8 + (threadIdx.x >> 5);
  __shared__ float sh_pp[30][8];          // window-projection partials
  __shared__ float sh_p[30];              // p = h1 @ Wwin^T + bwin (this row)
  __shared__ float sh_al[10], sh_be[10], sh_kp[10], sh_phi[65];

  for (int t = 0; t < cT; ++t) {
    const unsigned short* hread = (t & 1) ? hb1p : hb0;
    unsigned short* hwrite      = (t & 1) ? hb0 : hb1p;

    // ---- stage A: gates GEMM (WMMA bf16) fused with LSTM nonlinearity ----
    if (gwave < 100) {
      const int mt = gwave & 3, ntg = gwave >> 2;
      f32x8 z = {};
      f32x8 acc[4] = { z, z, z, z };
      #pragma unroll
      for (int kt = 0; kt < cKT1; ++kt) {
        bf16x16 a = load_afrag(inp1, cIN1P, mt * 16, kt, lane);
        #pragma unroll
        for (int gc = 0; gc < 4; ++gc)
          acc[gc] = wmma_bf16(a, load_bfrag(Wih1p, (gc * 25 + ntg) * cKT1 + kt, lane), acc[gc]);
      }
      for (int kt = 0; kt < cKTH; ++kt) {
        bf16x16 a = load_afrag(hread, cHP, mt * 16, kt, lane);
        #pragma unroll
        for (int gc = 0; gc < 4; ++gc)
          acc[gc] = wmma_bf16(a, load_bfrag(Whh1p, (gc * 25 + ntg) * cKTH + kt, lane), acc[gc]);
      }
      const int n = lane & 15, mhi = (lane >> 4) << 3, j = ntg * 16 + n;
      #pragma unroll
      for (int r = 0; r < 8; ++r) {
        const int brow = mt * 16 + mhi + r;
        float gi = acc[0][r] + bih1[j]        + bhh1[j];
        float gf = acc[1][r] + bih1[j + 400]  + bhh1[j + 400];
        float gg = acc[2][r] + bih1[j + 800]  + bhh1[j + 800];
        float go = acc[3][r] + bih1[j + 1200] + bhh1[j + 1200];
        float cp = c1s[brow * cH + j];
        float cn = sigm(gf) * cp + sigm(gi) * tanhf(gg);
        float hn = sigm(go) * tanhf(cn);
        c1s[brow * cH + j] = cn;
        h1s[brow * cH + j] = hn;
        unsigned short hb = f2bf(hn);
        hwrite[brow * cHP + j] = hb;
        const size_t R = (size_t)t * cB + brow;
        cat[R * cCATP + j] = hb;
        in2[R * cIN2P + 3 + j] = hb;
      }
    }
    grid_barrier(bar, NBLK);

    // ---- stage B: window projection + attention, one batch row per block-iter ----
    for (int b = blockIdx.x; b < cB; b += NBLK) {
      const int tid = threadIdx.x;
      // p[b,col] split 8 ways along K (400 = 8 x 50), reduced in LDS
      if (tid < 240) {
        const int col = tid >> 3, part = tid & 7;
        float s = 0.f;
        const float* hr = h1s + b * cH + part * 50;
        const float* wr = Wwin + col * cH + part * 50;
        for (int k = 0; k < 50; ++k) s += hr[k] * wr[k];
        sh_pp[col][part] = s;
      }
      __syncthreads();
      if (tid < 30) {
        float s = bwin[tid];
        #pragma unroll
        for (int i = 0; i < 8; ++i) s += sh_pp[tid][i];
        sh_p[tid] = s;
      }
      __syncthreads();
      if (tid < cK) {
        float kp = kappa[(t & 1) * 640 + b * cK + tid];
        float kn = kp + expf(sh_p[20 + tid]);
        kappa[((t + 1) & 1) * 640 + b * cK + tid] = kn;
        sh_al[tid] = expf(sh_p[tid]);
        sh_be[tid] = expf(sh_p[10 + tid]);
        sh_kp[tid] = kn;
      }
      __syncthreads();
      if (tid < cS + 1) {
        float u = (float)tid, s = 0.f;
        for (int k = 0; k < cK; ++k) {
          float d = sh_kp[k] - u;
          s += sh_al[k] * expf(-sh_be[k] * d * d);
        }
        s *= (float)cS / sentlen[b];
        sh_phi[tid] = s;
        phibuf[b * (cS + 1) + tid] = s;
      }
      __syncthreads();
      if (tid < cSD) {
        float s = 0.f;
        for (int si = 0; si < cS; ++si) s += sh_phi[si] * onehots[((size_t)b * cS + si) * cSD + tid];
        wbuf[b * cSD + tid] = s;
        unsigned short wv = f2bf(s);
        const size_t R = (size_t)t * cB + b;
        in2[R * cIN2P + 403 + tid] = wv;
        in3[R * cIN2P + 403 + tid] = wv;
        if (t + 1 < cT) inp1[b * 64 + 3 + tid] = wv;
      }
      if (tid < 3 && t + 1 < cT)
        inp1[b * 64 + tid] = f2bf(strks[((size_t)b * cT + (t + 1)) * 3 + tid]);
      __syncthreads();
    }
    grid_barrier(bar, NBLK);
  }
}

// ---------------- phases 2/3: plain LSTM scan (persistent, 1 barrier/step) ----------------
__global__ __launch_bounds__(256) void hs_lstm(
    const unsigned short* __restrict__ inbuf,   // [T][64][480] bf16
    const unsigned short* __restrict__ Wihp,    // [100][15][512]
    const unsigned short* __restrict__ Whhp,    // [100][13][512]
    const float* __restrict__ bih, const float* __restrict__ bhh,
    float* __restrict__ cs, float* __restrict__ hs32,
    unsigned short* __restrict__ hb0, unsigned short* __restrict__ hb1p,
    unsigned short* __restrict__ cat, int colbase,
    unsigned short* __restrict__ dest2,         // nullable: next layer input (+3 cols)
    unsigned* bar) {
  const int lane  = threadIdx.x & 31;
  const int gwave = blockIdx.x * 8 + (threadIdx.x >> 5);

  for (int t = 0; t < cT; ++t) {
    const unsigned short* hread = (t & 1) ? hb1p : hb0;
    unsigned short* hwrite      = (t & 1) ? hb0 : hb1p;
    if (gwave < 100) {
      const int mt = gwave & 3, ntg = gwave >> 2;
      const unsigned short* inrow = inbuf + (size_t)t * cB * cIN2P;
      f32x8 z = {};
      f32x8 acc[4] = { z, z, z, z };
      for (int kt = 0; kt < cKT2; ++kt) {
        bf16x16 a = load_afrag(inrow, cIN2P, mt * 16, kt, lane);
        #pragma unroll
        for (int gc = 0; gc < 4; ++gc)
          acc[gc] = wmma_bf16(a, load_bfrag(Wihp, (gc * 25 + ntg) * cKT2 + kt, lane), acc[gc]);
      }
      for (int kt = 0; kt < cKTH; ++kt) {
        bf16x16 a = load_afrag(hread, cHP, mt * 16, kt, lane);
        #pragma unroll
        for (int gc = 0; gc < 4; ++gc)
          acc[gc] = wmma_bf16(a, load_bfrag(Whhp, (gc * 25 + ntg) * cKTH + kt, lane), acc[gc]);
      }
      const int n = lane & 15, mhi = (lane >> 4) << 3, j = ntg * 16 + n;
      #pragma unroll
      for (int r = 0; r < 8; ++r) {
        const int brow = mt * 16 + mhi + r;
        float gi = acc[0][r] + bih[j]        + bhh[j];
        float gf = acc[1][r] + bih[j + 400]  + bhh[j + 400];
        float gg = acc[2][r] + bih[j + 800]  + bhh[j + 800];
        float go = acc[3][r] + bih[j + 1200] + bhh[j + 1200];
        float cp = cs[brow * cH + j];
        float cn = sigm(gf) * cp + sigm(gi) * tanhf(gg);
        float hn = sigm(go) * tanhf(cn);
        cs[brow * cH + j] = cn;
        hs32[brow * cH + j] = hn;
        unsigned short hbv = f2bf(hn);
        hwrite[brow * cHP + j] = hbv;
        const size_t R = (size_t)t * cB + brow;
        cat[R * cCATP + colbase + j] = hbv;
        if (dest2) dest2[R * cIN2P + 3 + j] = hbv;
      }
    }
    grid_barrier(bar, NBLK);
  }
}

// ---------------- MDN head GEMM: [38400 x 1216] @ [1216 x 128] bf16 WMMA ----------------
__global__ __launch_bounds__(256) void hs_mdn(
    const unsigned short* __restrict__ cat, const unsigned short* __restrict__ Wp,
    float* __restrict__ params) {
  const int lane = threadIdx.x & 31;
  const int nt   = threadIdx.x >> 5;     // 0..7 (128/16 n-tiles)
  const int mtile = blockIdx.x;          // 0..2399
  f32x8 acc = {};
  for (int kt = 0; kt < cKTC; ++kt) {
    bf16x16 a = load_afrag(cat, cCATP, mtile * 16, kt, lane);
    bf16x16 b = load_bfrag(Wp, nt * cKTC + kt, lane);
    acc = wmma_bf16(a, b, acc);
  }
  const int n = lane & 15, mhi = (lane >> 4) << 3;
  #pragma unroll
  for (int r = 0; r < 8; ++r) {
    const int R = mtile * 16 + mhi + r;
    params[(size_t)R * 128 + nt * 16 + n] = acc[r];
  }
}

// ---------------- finalize: MDN nonlinearities into d_out ----------------
__global__ __launch_bounds__(256) void hs_finalize(
    const float* __restrict__ params, const float* __restrict__ bmdn,
    float* __restrict__ out) {
  const int q = blockIdx.x * 256 + threadIdx.x;     // q = b*T + t
  if (q >= cE) return;
  const int b = q / cT, t = q % cT;
  const float* P = params + (size_t)(t * cB + b) * 128;
  out[q] = sigm(P[0] + bmdn[0]);                                   // eos
  float wh[cMIX]; float mx = -3.0e38f;
  for (int m = 0; m < cMIX; ++m) { wh[m] = P[1 + m] + bmdn[1 + m]; mx = fmaxf(mx, wh[m]); }
  float ssum = 0.f;
  for (int m = 0; m < cMIX; ++m) { float e = expf(wh[m] - mx); wh[m] = e; ssum += e; }
  for (int m = 0; m < cMIX; ++m) out[(size_t)cE + (size_t)q * cMIX + m] = wh[m] / ssum;  // weights
  for (int m = 0; m < cMIX; ++m) out[21ull * cE + (size_t)q * cMIX + m] = P[21 + m] + bmdn[21 + m];  // mu1
  for (int m = 0; m < cMIX; ++m) out[41ull * cE + (size_t)q * cMIX + m] = P[41 + m] + bmdn[41 + m];  // mu2
  for (int m = 0; m < cMIX; ++m) out[61ull * cE + (size_t)q * cMIX + m] = expf(P[61 + m] + bmdn[61 + m]);  // sigma1
  for (int m = 0; m < cMIX; ++m) out[81ull * cE + (size_t)q * cMIX + m] = expf(P[81 + m] + bmdn[81 + m]);  // sigma2
  for (int m = 0; m < cMIX; ++m) out[101ull * cE + (size_t)q * cMIX + m] = tanhf(P[101 + m] + bmdn[101 + m]);  // rho
}

// ---------------- tails: final-state outputs ----------------
__global__ __launch_bounds__(256) void hs_tails(char* ws, float* __restrict__ out) {
  const int gid = blockIdx.x * 256 + threadIdx.x;
  const size_t base = 121ull * cE;
  const float* wbuf = (const float*)(ws + O_WBUF);
  const float* kap  = (const float*)(ws + O_KAPPA);   // parity (T&1)==0 -> first half
  const float* h1s  = (const float*)(ws + O_H1S);
  const float* c1s  = (const float*)(ws + O_C1);
  const float* h2s  = (const float*)(ws + O_H2S);
  const float* c2s  = (const float*)(ws + O_C2);
  const float* h3s  = (const float*)(ws + O_H3S);
  const float* c3s  = (const float*)(ws + O_C3);
  const float* phib = (const float*)(ws + O_PHI);
  if      (gid < 3840)   out[base + gid] = wbuf[gid];
  else if (gid < 4480)   out[base + gid] = kap[gid - 3840];
  else if (gid < 30080)  out[base + gid] = h1s[gid - 4480];
  else if (gid < 55680)  out[base + gid] = c1s[gid - 30080];
  else if (gid < 81280)  out[base + gid] = h2s[gid - 55680];
  else if (gid < 106880) out[base + gid] = c2s[gid - 81280];
  else if (gid < 132480) out[base + gid] = h3s[gid - 106880];
  else if (gid < 158080) out[base + gid] = c3s[gid - 132480];
  else if (gid < 162240) out[base + gid] = phib[gid - 158080];
}

// ---------------- host launch ----------------
extern "C" void kernel_launch(void* const* d_in, const int* in_sizes, int n_in,
                              void* d_out, int out_size, void* d_ws, size_t ws_size,
                              hipStream_t stream) {
  (void)in_sizes; (void)n_in; (void)out_size; (void)ws_size;
  const float* strks   = (const float*)d_in[0];
  const float* onehots = (const float*)d_in[2];
  const float* sents_m = (const float*)d_in[3];
  const float* w_prev  = (const float*)d_in[4];
  const float* k_prev  = (const float*)d_in[5];
  const float* h1 = (const float*)d_in[6];  const float* c1 = (const float*)d_in[7];
  const float* h2 = (const float*)d_in[8];  const float* c2 = (const float*)d_in[9];
  const float* h3 = (const float*)d_in[10]; const float* c3 = (const float*)d_in[11];
  const float* W_ih1 = (const float*)d_in[12]; const float* W_hh1 = (const float*)d_in[13];
  const float* b_ih1 = (const float*)d_in[14]; const float* b_hh1 = (const float*)d_in[15];
  const float* W_win = (const float*)d_in[16]; const float* b_win = (const float*)d_in[17];
  const float* W_ih2 = (const float*)d_in[18]; const float* W_hh2 = (const float*)d_in[19];
  const float* b_ih2 = (const float*)d_in[20]; const float* b_hh2 = (const float*)d_in[21];
  const float* W_ih3 = (const float*)d_in[22]; const float* W_hh3 = (const float*)d_in[23];
  const float* b_ih3 = (const float*)d_in[24]; const float* b_hh3 = (const float*)d_in[25];
  const float* W_mdn = (const float*)d_in[26]; const float* b_mdn = (const float*)d_in[27];
  char* ws = (char*)d_ws;
  float* out = (float*)d_out;

  hs_init<<<150, 256, 0, stream>>>(strks, sents_m, w_prev, k_prev, h1, c1, h2, c2, h3, c3, ws);

  hs_pack<<<400,  256, 0, stream>>>(W_ih1, (unsigned short*)(ws + O_WIH1P), cG, 63,  cNT, cKT1);
  hs_pack<<<2600, 256, 0, stream>>>(W_hh1, (unsigned short*)(ws + O_WHH1P), cG, cH,  cNT, cKTH);
  hs_pack<<<3000, 256, 0, stream>>>(W_ih2, (unsigned short*)(ws + O_WIH2P), cG, 463, cNT, cKT2);
  hs_pack<<<2600, 256, 0, stream>>>(W_hh2, (unsigned short*)(ws + O_WHH2P), cG, cH,  cNT, cKTH);
  hs_pack<<<3000, 256, 0, stream>>>(W_ih3, (unsigned short*)(ws + O_WIH3P), cG, 463, cNT, cKT2);
  hs_pack<<<2600, 256, 0, stream>>>(W_hh3, (unsigned short*)(ws + O_WHH3P), cG, cH,  cNT, cKTH);
  hs_pack<<<608,  256, 0, stream>>>(W_mdn, (unsigned short*)(ws + O_WMDNP), 121, 1200, 8, cKTC);

  hs_phase1<<<NBLK, 256, 0, stream>>>(strks, onehots, b_ih1, b_hh1, W_win, b_win, ws);

  hs_lstm<<<NBLK, 256, 0, stream>>>(
      (const unsigned short*)(ws + O_IN2),
      (const unsigned short*)(ws + O_WIH2P), (const unsigned short*)(ws + O_WHH2P),
      b_ih2, b_hh2, (float*)(ws + O_C2), (float*)(ws + O_H2S),
      (unsigned short*)(ws + O_HB2), (unsigned short*)(ws + O_HB2) + 26624,
      (unsigned short*)(ws + O_CAT), 400,
      (unsigned short*)(ws + O_IN3), (unsigned*)(ws + O_BAR) + 1);

  hs_lstm<<<NBLK, 256, 0, stream>>>(
      (const unsigned short*)(ws + O_IN3),
      (const unsigned short*)(ws + O_WIH3P), (const unsigned short*)(ws + O_WHH3P),
      b_ih3, b_hh3, (float*)(ws + O_C3), (float*)(ws + O_H3S),
      (unsigned short*)(ws + O_HB3), (unsigned short*)(ws + O_HB3) + 26624,
      (unsigned short*)(ws + O_CAT), 800,
      (unsigned short*)nullptr, (unsigned*)(ws + O_BAR) + 2);

  hs_mdn<<<2400, 256, 0, stream>>>((const unsigned short*)(ws + O_CAT),
                                   (const unsigned short*)(ws + O_WMDNP),
                                   (float*)(ws + O_PAR));
  hs_finalize<<<150, 256, 0, stream>>>((const float*)(ws + O_PAR), b_mdn, out);
  hs_tails<<<634, 256, 0, stream>>>(ws, out);
}
